// HierMatcher_1202590843225
// MI455X (gfx1250) — matrix operand order repaired and checked by reading, not compiled, not run on hardware
//
#include <hip/hip_runtime.h>
#include <hip/hip_bf16.h>
#include <math.h>

typedef _Float16 f16;
typedef __attribute__((ext_vector_type(16))) _Float16 v16h;
typedef __attribute__((ext_vector_type(8)))  _Float16 v8h;
typedef __attribute__((ext_vector_type(8)))  float    v8f;
typedef __attribute__((ext_vector_type(8)))  short    s8s;

#define D    768
#define NTOK 512
#define LROW 776      // padded f16 row pitch: 1552B = 388 dwords, 388 % 64 = 4 -> conflict-free
#define NATTR 8
#define DE   12288    // (8+8)*768

// ---------------------------------------------------------------------------
// |x| on packed f16 via sign-bit mask
static __device__ __forceinline__ v8h habs8(v8h x) {
  union { v8h h; s8s s; } u; u.h = x;
  u.s = u.s & (short)0x7fff;
  return u.h;
}

static __device__ __forceinline__ v16h cat16(v8h lo, v8h hi) {
  return __builtin_shufflevector(lo, hi, 0,1,2,3,4,5,6,7,8,9,10,11,12,13,14,15);
}

// ---------------------------------------------------------------------------
// Kernel 1: fp32 -> f16 transpose of the two token-highway weight matrices.
// WnT[d*768 + k] = Wn[k*768 + d]  so a WMMA B-fragment (two runs of 8 K for a
// fixed output dim d) is two contiguous 16B loads.
__global__ __launch_bounds__(256) void convert_weights(
    const float* __restrict__ Wn, const float* __restrict__ Wg,
    f16* __restrict__ WnT, f16* __restrict__ WgT) {
  int idx = blockIdx.x * 256 + threadIdx.x;
  if (idx >= D * D) return;
  int dd = idx / D, k = idx - dd * D;
  WnT[idx] = (f16)Wn[k * D + dd];
  WgT[idx] = (f16)Wg[k * D + dd];
}

// ---------------------------------------------------------------------------
// Kernel 2: fused token-matching score matrix.
// scores[i,j] = sum_d ( sig(gpre_d)*relu(npre_d) + (1-sig(gpre_d))*cmp_d ) * Wl_d
// with cmp = |left_i - right_j|, npre = cmp@Wn + bn, gpre = cmp@Wg + bg.
// Block = 8 waves = 16 i x 16 j pair tile; wave w owns j = jbase+2w, jbase+2w+1
// (two WMMA M-tiles of pairs sharing every B fragment) and loops all 48
// output-dim chunks: per K-step 6 LDS + 4 global b128 loads feed 4 WMMAs.
__global__ __launch_bounds__(256) void score_kernel(
    const float* __restrict__ left, const float* __restrict__ right,
    const f16* __restrict__ WnT, const f16* __restrict__ WgT,
    const float* __restrict__ bn, const float* __restrict__ bg,
    const float* __restrict__ Wl, float* __restrict__ scores) {
  __shared__ __align__(16) f16 sl[16 * LROW];
  __shared__ __align__(16) f16 sr[16 * LROW];
  const int ibase = blockIdx.x * 16;
  const int jbase = blockIdx.y * 16;
  const int tid = threadIdx.x;

  // stage embedding tiles to LDS as f16
  for (int idx = tid; idx < 16 * D; idx += 256) {
    int r = idx / D, d = idx - r * D;
    sl[r * LROW + d] = (f16)left[(ibase + r) * D + d];
    sr[r * LROW + d] = (f16)right[(jbase + r) * D + d];
  }
  __syncthreads();

  const int w    = tid >> 5;      // wave id; owns local j = 2w, 2w+1
  const int lane = tid & 31;
  const int hh   = lane >> 4;     // lane half (K-phase for A/B, M-offset for C/D)
  const int lcol = lane & 15;     // A row / B,C,D column
  const int kb   = hh ? 8 : 0;    // A-fragment K base per ISA 16-bit A layout

  const f16* lrow  = sl + lcol * LROW;
  const f16* rrow0 = sr + (2 * w) * LROW;
  const f16* rrow1 = sr + (2 * w + 1) * LROW;

  float sacc0[8], sacc1[8];
  #pragma unroll
  for (int r = 0; r < 8; ++r) { sacc0[r] = 0.f; sacc1[r] = 0.f; }

  for (int c = 0; c < D / 16; ++c) {
    const int dcol = c * 16 + lcol;
    v8f accn0 = {}, accg0 = {}, accn1 = {}, accg1 = {};
    const f16* bnp = WnT + (size_t)dcol * D + hh * 8;
    const f16* bgp = WgT + (size_t)dcol * D + hh * 8;
    #pragma unroll 2
    for (int k0 = 0; k0 < D; k0 += 32) {
      // shared left fragment + two right fragments -> two A fragments
      v8h llo = *(const v8h*)(lrow + k0 + kb);
      v8h lhi = *(const v8h*)(lrow + k0 + 16 + kb);
      v8h r0l = *(const v8h*)(rrow0 + k0 + kb);
      v8h r0h = *(const v8h*)(rrow0 + k0 + 16 + kb);
      v8h r1l = *(const v8h*)(rrow1 + k0 + kb);
      v8h r1h = *(const v8h*)(rrow1 + k0 + 16 + kb);
      v16h A0 = cat16(habs8(llo - r0l), habs8(lhi - r0h));
      v16h A1 = cat16(habs8(llo - r1l), habs8(lhi - r1h));
      // B fragments: K in ISA table order (8h+s, 16+8h+s) of transposed weights, shared by A0/A1
      v16h Bn = cat16(*(const v8h*)(bnp + k0), *(const v8h*)(bnp + k0 + 16));
      v16h Bg = cat16(*(const v8h*)(bgp + k0), *(const v8h*)(bgp + k0 + 16));
      accn0 = __builtin_amdgcn_wmma_f32_16x16x32_f16(
                  false, A0, false, Bn, (short)0, accn0, false, false);
      accg0 = __builtin_amdgcn_wmma_f32_16x16x32_f16(
                  false, A0, false, Bg, (short)0, accg0, false, false);
      accn1 = __builtin_amdgcn_wmma_f32_16x16x32_f16(
                  false, A1, false, Bn, (short)0, accn1, false, false);
      accg1 = __builtin_amdgcn_wmma_f32_16x16x32_f16(
                  false, A1, false, Bg, (short)0, accg1, false, false);
    }
    // fused highway elementwise + Wl projection for this 16-dim chunk.
    // C/D layout: acc[r] -> pair row (r + 8*hh), out dim = dcol.
    const float bnd = bn[dcol], bgd = bg[dcol], wld = Wl[dcol];
    const float rv0 = (float)rrow0[dcol];
    const float rv1 = (float)rrow1[dcol];
    #pragma unroll
    for (int r = 0; r < 8; ++r) {
      const float lv = (float)sl[(r + 8 * hh) * LROW + dcol];
      {
        float cv = fabsf(lv - rv0);
        float nv = accn0[r] + bnd; nv = nv > 0.f ? nv : 0.f;
        float gv = 1.f / (1.f + __expf(-(accg0[r] + bgd)));
        sacc0[r] += (gv * nv + (1.f - gv) * cv) * wld;
      }
      {
        float cv = fabsf(lv - rv1);
        float nv = accn1[r] + bnd; nv = nv > 0.f ? nv : 0.f;
        float gv = 1.f / (1.f + __expf(-(accg1[r] + bgd)));
        sacc1[r] += (gv * nv + (1.f - gv) * cv) * wld;
      }
    }
  }
  // reduce score partials over the 16 lanes of each half (same pairs, diff d)
  #pragma unroll
  for (int m = 1; m < 16; m <<= 1) {
    #pragma unroll
    for (int r = 0; r < 8; ++r) {
      sacc0[r] += __shfl_xor(sacc0[r], m, 32);
      sacc1[r] += __shfl_xor(sacc1[r], m, 32);
    }
  }
  if (lcol == 0) {
    #pragma unroll
    for (int r = 0; r < 8; ++r) {
      int i = ibase + r + 8 * hh;
      scores[i * NTOK + jbase + 2 * w]     = sacc0[r];
      scores[i * NTOK + jbase + 2 * w + 1] = sacc1[r];
    }
  }
}

// ---------------------------------------------------------------------------
// Kernel 3: argmax over rows (left matches) and cols (right matches), with
// jnp.argmax first-max tie-break.
__global__ __launch_bounds__(256) void argmax_kernel(
    const float* __restrict__ scores, int* __restrict__ lidx, int* __restrict__ ridx) {
  __shared__ float smax[256];
  __shared__ int   sidx[256];
  const int id  = blockIdx.x;          // 0..511 rows, 512..1023 cols
  const int tid = threadIdx.x;
  const bool row = id < NTOK;
  const int base = row ? id : (id - NTOK);
  float bmax = -__builtin_inff();
  int   bidx = 0x7fffffff;
  for (int t = tid; t < NTOK; t += 256) {
    float v = row ? scores[base * NTOK + t] : scores[t * NTOK + base];
    if (v > bmax || (v == bmax && t < bidx)) { bmax = v; bidx = t; }
  }
  smax[tid] = bmax; sidx[tid] = bidx;
  __syncthreads();
  for (int s = 128; s > 0; s >>= 1) {
    if (tid < s) {
      float v = smax[tid + s]; int ix = sidx[tid + s];
      if (v > smax[tid] || (v == smax[tid] && ix < sidx[tid])) {
        smax[tid] = v; sidx[tid] = ix;
      }
    }
    __syncthreads();
  }
  if (tid == 0) { if (row) lidx[base] = sidx[0]; else ridx[base] = sidx[0]; }
}

// ---------------------------------------------------------------------------
// Kernel 4: attribute matching (ragged softmax + weighted sum of matched cmp
// rows). One block per attribute: 0..7 left, 8..15 right. Writes the entity
// concat vector xcat[16*768] directly.
__global__ __launch_bounds__(256) void attr_kernel(
    const float* __restrict__ left, const float* __restrict__ right,
    const int* __restrict__ llen, const int* __restrict__ rlen,
    const int* __restrict__ lidx, const int* __restrict__ ridx,
    const float* __restrict__ attrL, const float* __restrict__ attrR,
    const float* __restrict__ empty_res, float* __restrict__ xcat) {
  __shared__ float slog[NTOK];
  __shared__ float red[256];
  const int a   = blockIdx.x;
  const bool isL = a < NATTR;
  const int ai  = isL ? a : a - NATTR;
  const int* lens = isL ? llen : rlen;
  const float* emb   = isL ? left  : right;
  const float* other = isL ? right : left;
  const int* midx = isL ? lidx : ridx;
  const float* fe = (isL ? attrL : attrR) + ai * D;
  float* out = xcat + a * D;
  const int tid = threadIdx.x;

  int start = 0;
  for (int b = 0; b < ai; ++b) start += lens[b];
  const int len = lens[ai];

  if (len == 0) {
    for (int d = tid; d < D; d += 256) out[d] = empty_res[d];
    return;
  }
  // logits[t] = <emb_t, field_emb[a]>
  for (int t = tid; t < len; t += 256) {
    const float* e = emb + (size_t)(start + t) * D;
    float acc = 0.f;
    for (int d = 0; d < D; ++d) acc += e[d] * fe[d];
    slog[t] = acc;
  }
  __syncthreads();
  float m = -__builtin_inff();
  for (int t = tid; t < len; t += 256) m = fmaxf(m, slog[t]);
  red[tid] = m; __syncthreads();
  for (int s = 128; s > 0; s >>= 1) {
    if (tid < s) red[tid] = fmaxf(red[tid], red[tid + s]);
    __syncthreads();
  }
  m = red[0]; __syncthreads();
  float sum = 0.f;
  for (int t = tid; t < len; t += 256) {
    float e = __expf(slog[t] - m);
    slog[t] = e;
    sum += e;
  }
  red[tid] = sum; __syncthreads();
  for (int s = 128; s > 0; s >>= 1) {
    if (tid < s) red[tid] += red[tid + s];
    __syncthreads();
  }
  const float inv_denom = 1.f / red[0];
  __syncthreads();
  // rep[d] = sum_t w_t * |emb[t,d] - other[match(t),d]|
  for (int d = tid; d < D; d += 256) {
    float acc = 0.f;
    for (int t = 0; t < len; ++t) {
      int tok = start + t;
      float c = fabsf(emb[(size_t)tok * D + d] - other[(size_t)midx[tok] * D + d]);
      acc += slog[t] * c;
    }
    out[d] = acc * inv_denom;
  }
}

// ---------------------------------------------------------------------------
// Kernel 5: entity highway, HBM-bound GEMV over two 12288x12288 fp32 matrices.
// x is broadcast from LDS; column reads are coalesced; each matrix is streamed
// from HBM exactly once (~1.2GB total -> ~52us at 23.3 TB/s).
__global__ __launch_bounds__(256) void ent_highway(
    const float* __restrict__ x,
    const float* __restrict__ Wn, const float* __restrict__ bn,
    const float* __restrict__ Wg, const float* __restrict__ bg,
    float* __restrict__ hwv) {
  __shared__ float sx[DE];   // 48KB
  const int tid = threadIdx.x;
  for (int k = tid; k < DE; k += 256) sx[k] = x[k];
  __syncthreads();
  const int d = blockIdx.x * 256 + tid;
  float an = 0.f, ag = 0.f;
  #pragma unroll 4
  for (int k = 0; k < DE; ++k) {
    float xv = sx[k];
    an += xv * Wn[(size_t)k * DE + d];
    ag += xv * Wg[(size_t)k * DE + d];
  }
  float nv = an + bn[d];
  nv = nv > 0.f ? nv : 0.f;
  float gv = 1.f / (1.f + __expf(-(ag + bg[d])));
  hwv[d] = gv * nv + (1.f - gv) * x[d];
}

// ---------------------------------------------------------------------------
// Kernel 6: final 12288x2 projection + softmax.
__global__ __launch_bounds__(256) void ent_final(
    const float* __restrict__ hwv, const float* __restrict__ Wl,
    const float* __restrict__ bl, float* __restrict__ out) {
  __shared__ float r0[256], r1[256];
  const int tid = threadIdx.x;
  float a0 = 0.f, a1 = 0.f;
  for (int d = tid; d < DE; d += 256) {
    float h = hwv[d];
    a0 += h * Wl[d * 2 + 0];
    a1 += h * Wl[d * 2 + 1];
  }
  r0[tid] = a0; r1[tid] = a1;
  __syncthreads();
  for (int s = 128; s > 0; s >>= 1) {
    if (tid < s) { r0[tid] += r0[tid + s]; r1[tid] += r1[tid + s]; }
    __syncthreads();
  }
  if (tid == 0) {
    float l0 = r0[0] + bl[0], l1 = r1[0] + bl[1];
    float mx = fmaxf(l0, l1);
    float e0 = __expf(l0 - mx), e1 = __expf(l1 - mx);
    float inv = 1.f / (e0 + e1);
    out[0] = e0 * inv;
    out[1] = e1 * inv;
  }
}

// ---------------------------------------------------------------------------
// Workspace layout (bytes)
#define WS_WNT    0u
#define WS_WGT    1179648u                 // 768*768*2
#define WS_SCORES 2359296u                 // + 768*768*2
#define WS_LIDX   3407872u                 // + 512*512*4
#define WS_RIDX   3409920u                 // + 2048
#define WS_XCAT   3411968u                 // + 2048
#define WS_HWV    3461120u                 // + 12288*4

extern "C" void kernel_launch(void* const* d_in, const int* in_sizes, int n_in,
                              void* d_out, int out_size, void* d_ws, size_t ws_size,
                              hipStream_t stream) {
  const float* left   = (const float*)d_in[0];
  const float* right  = (const float*)d_in[1];
  const int*   llen   = (const int*)d_in[2];
  const int*   rlen   = (const int*)d_in[3];
  const float* Wn     = (const float*)d_in[4];
  const float* bn     = (const float*)d_in[5];
  const float* Wg     = (const float*)d_in[6];
  const float* bg     = (const float*)d_in[7];
  const float* Wl     = (const float*)d_in[8];
  // d_in[9] lin_tok_b: constant shift, argmax-invariant -> unused
  const float* attrL  = (const float*)d_in[10];
  const float* attrR  = (const float*)d_in[11];
  const float* WnE    = (const float*)d_in[12];
  const float* bnE    = (const float*)d_in[13];
  const float* WgE    = (const float*)d_in[14];
  const float* bgE    = (const float*)d_in[15];
  const float* WlE    = (const float*)d_in[16];
  const float* blE    = (const float*)d_in[17];
  const float* emptyr = (const float*)d_in[18];

  char* ws = (char*)d_ws;
  f16*   WnT    = (f16*)(ws + WS_WNT);
  f16*   WgT    = (f16*)(ws + WS_WGT);
  float* scores = (float*)(ws + WS_SCORES);
  int*   lidx   = (int*)(ws + WS_LIDX);
  int*   ridx   = (int*)(ws + WS_RIDX);
  float* xcat   = (float*)(ws + WS_XCAT);
  float* hwv    = (float*)(ws + WS_HWV);

  convert_weights<<<(D * D + 255) / 256, 256, 0, stream>>>(Wn, Wg, WnT, WgT);
  score_kernel<<<dim3(NTOK / 16, NTOK / 16), 256, 0, stream>>>(
      left, right, WnT, WgT, bn, bg, Wl, scores);
  argmax_kernel<<<2 * NTOK, 256, 0, stream>>>(scores, lidx, ridx);
  attr_kernel<<<2 * NATTR, 256, 0, stream>>>(
      left, right, llen, rlen, lidx, ridx, attrL, attrR, emptyr, xcat);
  ent_highway<<<DE / 256, 256, 0, stream>>>(xcat, WnE, bnE, WgE, bgE, hwv);
  ent_final<<<1, 256, 0, stream>>>(hwv, WlE, blE, (float*)d_out);
}
